// MoleculeDecoder_55679956025713
// MI455X (gfx1250) — matrix operand order repaired and checked by reading, not compiled, not run on hardware
//
#include <hip/hip_runtime.h>
#include <hip/hip_bf16.h>
#include <math.h>

typedef __attribute__((ext_vector_type(16))) __bf16 v16bf;
typedef __attribute__((ext_vector_type(8)))  __bf16 v8bf;
typedef __attribute__((ext_vector_type(8)))  float  v8f;

#define SHUF16(lo, hi) __builtin_shufflevector((lo), (hi), 0,1,2,3,4,5,6,7,8,9,10,11,12,13,14,15)

static constexpr int Bsz = 32;
static constexpr int E   = 768;
static constexpr int V   = 8192;
static constexpr int T   = 256;
static constexpr int G3  = 3 * E;    // 2304

// A-fragment: 16-bit A 16x32, this code's own K order (same as B), lane = row, K = kh*16 + 0..15 (contiguous).
// p = row_base + kb + kh*16  ->  halves at p and p+8.
__device__ __forceinline__ v16bf loadA16(const __bf16* p) {
    return SHUF16(*(const v8bf*)p, *(const v8bf*)(p + 8));
}
// B-fragment: 16-bit B 32x16, this code's own K order (same as A), lane = col, K = kh*16 + 0..15 (contiguous).
// p = col_row_base + kb + kh*16 -> halves at p and p+8.
__device__ __forceinline__ v16bf loadB16(const __bf16* p) {
    return SHUF16(*(const v8bf*)p, *(const v8bf*)(p + 8));
}

// ---------------- f32 -> bf16 conversion (one-time per launch) ----------------
__global__ void cvt_bf16_kernel(const float* __restrict__ src, __bf16* __restrict__ dst, int n) {
    int i = blockIdx.x * blockDim.x + threadIdx.x;
    if (i < n) dst[i] = (__bf16)src[i];
}

// ---------------- h0 = latent @ W_lat^T + b_lat ; tok = 1 ----------------
__global__ void h0_kernel(const float* __restrict__ latent, const float* __restrict__ W_lat,
                          const float* __restrict__ b_lat,
                          float* __restrict__ h, __bf16* __restrict__ hbf, int* __restrict__ tok) {
    int gid = blockIdx.x * blockDim.x + threadIdx.x;          // 0..24575
    int b = gid / E, e = gid % E;
    const float* lr = latent + (size_t)b * E;
    const float* wr = W_lat + (size_t)e * E;
    float acc = 0.f;
    for (int k = 0; k < E; ++k) acc = fmaf(lr[k], wr[k], acc);
    acc += b_lat[e];
    h[gid]   = acc;
    hbf[gid] = (__bf16)acc;
    if (gid < Bsz) tok[gid] = 1;                              // <sos>
}

// ---------------- gi = emb[tok] @ W_ih^T + b_ih ; gh = h @ W_hh^T + b_hh ----------------
// Each wave: 1 M-tile, 2 N-tiles, both gate matmuls -> 4 WMMA chains, double-buffered.
// grid = (3E/16)/2 = 72, block = 64 (2 waves = 2 M tiles)
__global__ void __launch_bounds__(64, 1)
gates_kernel(const __bf16* __restrict__ embBf, const int* __restrict__ tok,
             const __bf16* __restrict__ hbf,
             const __bf16* __restrict__ WihBf, const __bf16* __restrict__ WhhBf,
             const float* __restrict__ b_ih, const float* __restrict__ b_hh,
             float* __restrict__ gi, float* __restrict__ gh) {
    const int lane  = threadIdx.x & 31;
    const int wave  = threadIdx.x >> 5;          // M tile (0..1)
    const int m     = lane & 15;
    const int kh    = lane >> 4;                 // K-half select

    const int arow = wave * 16 + m;              // batch row for A
    const int col0 = (blockIdx.x * 2 + 0) * 16 + m;
    const int col1 = (blockIdx.x * 2 + 1) * 16 + m;

    const __bf16* xp  = embBf + (size_t)tok[arow] * E + kh * 16;   // embedding gather
    const __bf16* hp  = hbf   + (size_t)arow * E + kh * 16;
    const __bf16* wi0 = WihBf + (size_t)col0 * E + kh * 16;
    const __bf16* wi1 = WihBf + (size_t)col1 * E + kh * 16;
    const __bf16* wh0 = WhhBf + (size_t)col0 * E + kh * 16;
    const __bf16* wh1 = WhhBf + (size_t)col1 * E + kh * 16;

    v8f ci0 = {}, ci1 = {}, ch0 = {}, ch1 = {};

    // prologue loads (generation 0)
    v16bf Ax  = loadA16(xp);
    v16bf Ah  = loadA16(hp);
    v16bf Bi0 = loadB16(wi0);
    v16bf Bi1 = loadB16(wi1);
    v16bf Bh0 = loadB16(wh0);
    v16bf Bh1 = loadB16(wh1);

    for (int kb = 32; kb < E; kb += 32) {
        // issue next generation's loads before consuming current generation
        v16bf Axn  = loadA16(xp + kb);
        v16bf Ahn  = loadA16(hp + kb);
        v16bf Bi0n = loadB16(wi0 + kb);
        v16bf Bi1n = loadB16(wi1 + kb);
        v16bf Bh0n = loadB16(wh0 + kb);
        v16bf Bh1n = loadB16(wh1 + kb);

        ci0 = __builtin_amdgcn_wmma_f32_16x16x32_bf16(false, Ax, false, Bi0, (short)0, ci0, false, false);
        ci1 = __builtin_amdgcn_wmma_f32_16x16x32_bf16(false, Ax, false, Bi1, (short)0, ci1, false, false);
        ch0 = __builtin_amdgcn_wmma_f32_16x16x32_bf16(false, Ah, false, Bh0, (short)0, ch0, false, false);
        ch1 = __builtin_amdgcn_wmma_f32_16x16x32_bf16(false, Ah, false, Bh1, (short)0, ch1, false, false);

        Ax = Axn; Ah = Ahn; Bi0 = Bi0n; Bi1 = Bi1n; Bh0 = Bh0n; Bh1 = Bh1n;
    }
    // epilogue: consume last generation
    ci0 = __builtin_amdgcn_wmma_f32_16x16x32_bf16(false, Ax, false, Bi0, (short)0, ci0, false, false);
    ci1 = __builtin_amdgcn_wmma_f32_16x16x32_bf16(false, Ax, false, Bi1, (short)0, ci1, false, false);
    ch0 = __builtin_amdgcn_wmma_f32_16x16x32_bf16(false, Ah, false, Bh0, (short)0, ch0, false, false);
    ch1 = __builtin_amdgcn_wmma_f32_16x16x32_bf16(false, Ah, false, Bh1, (short)0, ch1, false, false);

    const float bi0 = b_ih[col0], bi1 = b_ih[col1];
    const float bh0 = b_hh[col0], bh1 = b_hh[col1];
#pragma unroll
    for (int i = 0; i < 8; ++i) {
        int r = wave * 16 + kh * 8 + i;          // batch row of C element i
        gi[(size_t)r * G3 + col0] = ci0[i] + bi0;
        gi[(size_t)r * G3 + col1] = ci1[i] + bi1;
        gh[(size_t)r * G3 + col0] = ch0[i] + bh0;
        gh[(size_t)r * G3 + col1] = ch1[i] + bh1;
    }
}

// ---------------- GRU cell element-wise update ----------------
__global__ void update_kernel(const float* __restrict__ gi, const float* __restrict__ gh,
                              float* __restrict__ h, __bf16* __restrict__ hbf) {
    int gid = blockIdx.x * blockDim.x + threadIdx.x;          // 0..24575
    int b = gid / E, e = gid % E;
    const float* gib = gi + (size_t)b * G3;
    const float* ghb = gh + (size_t)b * G3;
    float ir = gib[e], iz = gib[E + e], in = gib[2 * E + e];
    float hr = ghb[e], hz = ghb[E + e], hn = ghb[2 * E + e];
    float r = 1.f / (1.f + __expf(-(ir + hr)));
    float z = 1.f / (1.f + __expf(-(iz + hz)));
    float n = tanhf(in + r * hn);
    float hv = h[gid];
    float hnew = (1.f - z) * n + z * hv;
    h[gid]   = hnew;
    hbf[gid] = (__bf16)hnew;
}

// ---------------- pred = h_new @ W_out^T + b_out -> d_out[b, t, :] ----------------
// Each wave: 1 M-tile, 4 N-tiles, double-buffered operand loads.
// grid = (V/16)/4 = 128, block = 64 (2 waves = 2 M tiles)
__global__ void __launch_bounds__(64, 1)
outmm_kernel(const __bf16* __restrict__ hbf, const __bf16* __restrict__ WoutBf,
             const float* __restrict__ b_out, float* __restrict__ out, int t) {
    const int lane  = threadIdx.x & 31;
    const int wave  = threadIdx.x >> 5;
    const int m     = lane & 15;
    const int kh    = lane >> 4;

    const int arow = wave * 16 + m;              // batch
    const int ntb  = blockIdx.x * 4;             // first of 4 N tiles

    const __bf16* Ap  = hbf + (size_t)arow * E + kh * 16;
    const __bf16* Bp0 = WoutBf + (size_t)((ntb + 0) * 16 + m) * E + kh * 16;
    const __bf16* Bp1 = WoutBf + (size_t)((ntb + 1) * 16 + m) * E + kh * 16;
    const __bf16* Bp2 = WoutBf + (size_t)((ntb + 2) * 16 + m) * E + kh * 16;
    const __bf16* Bp3 = WoutBf + (size_t)((ntb + 3) * 16 + m) * E + kh * 16;

    v8f a0 = {}, a1 = {}, a2 = {}, a3 = {};

    v16bf A  = loadA16(Ap);
    v16bf B0 = loadB16(Bp0);
    v16bf B1 = loadB16(Bp1);
    v16bf B2 = loadB16(Bp2);
    v16bf B3 = loadB16(Bp3);

    for (int kb = 32; kb < E; kb += 32) {
        v16bf An  = loadA16(Ap + kb);
        v16bf B0n = loadB16(Bp0 + kb);
        v16bf B1n = loadB16(Bp1 + kb);
        v16bf B2n = loadB16(Bp2 + kb);
        v16bf B3n = loadB16(Bp3 + kb);

        a0 = __builtin_amdgcn_wmma_f32_16x16x32_bf16(false, A, false, B0, (short)0, a0, false, false);
        a1 = __builtin_amdgcn_wmma_f32_16x16x32_bf16(false, A, false, B1, (short)0, a1, false, false);
        a2 = __builtin_amdgcn_wmma_f32_16x16x32_bf16(false, A, false, B2, (short)0, a2, false, false);
        a3 = __builtin_amdgcn_wmma_f32_16x16x32_bf16(false, A, false, B3, (short)0, a3, false, false);

        A = An; B0 = B0n; B1 = B1n; B2 = B2n; B3 = B3n;
    }
    a0 = __builtin_amdgcn_wmma_f32_16x16x32_bf16(false, A, false, B0, (short)0, a0, false, false);
    a1 = __builtin_amdgcn_wmma_f32_16x16x32_bf16(false, A, false, B1, (short)0, a1, false, false);
    a2 = __builtin_amdgcn_wmma_f32_16x16x32_bf16(false, A, false, B2, (short)0, a2, false, false);
    a3 = __builtin_amdgcn_wmma_f32_16x16x32_bf16(false, A, false, B3, (short)0, a3, false, false);

#pragma unroll
    for (int j = 0; j < 4; ++j) {
        const v8f& acc = (j == 0) ? a0 : (j == 1) ? a1 : (j == 2) ? a2 : a3;
        const int col = (ntb + j) * 16 + m;
        const float bias = b_out[col];
#pragma unroll
        for (int i = 0; i < 8; ++i) {
            int r = wave * 16 + kh * 8 + i;      // batch row
            out[(size_t)r * ((size_t)T * V) + (size_t)t * V + col] = acc[i] + bias;
        }
    }
}

// ---------------- greedy argmax over V per batch row -> tok ----------------
__global__ void argmax_kernel(const float* __restrict__ out, int t, int* __restrict__ tok) {
    const int b = blockIdx.x;
    const float* row = out + (size_t)b * ((size_t)T * V) + (size_t)t * V;
    __shared__ float sv[256];
    __shared__ int   si[256];
    float best = -INFINITY;
    int   bi   = 0;
    for (int v = threadIdx.x; v < V; v += 256) {
        float x = row[v];
        if (x > best) { best = x; bi = v; }
    }
    sv[threadIdx.x] = best;
    si[threadIdx.x] = bi;
    __syncthreads();
    for (int s = 128; s > 0; s >>= 1) {
        if (threadIdx.x < s) {
            float ov = sv[threadIdx.x + s];
            int   oi = si[threadIdx.x + s];
            if (ov > sv[threadIdx.x] || (ov == sv[threadIdx.x] && oi < si[threadIdx.x])) {
                sv[threadIdx.x] = ov;
                si[threadIdx.x] = oi;
            }
        }
        __syncthreads();
    }
    if (threadIdx.x == 0) tok[b] = si[0];
}

extern "C" void kernel_launch(void* const* d_in, const int* in_sizes, int n_in,
                              void* d_out, int out_size, void* d_ws, size_t ws_size,
                              hipStream_t stream) {
    const float* latent = (const float*)d_in[0];
    const float* W_lat  = (const float*)d_in[1];
    const float* b_lat  = (const float*)d_in[2];
    const float* emb    = (const float*)d_in[3];
    const float* W_ih   = (const float*)d_in[4];
    const float* W_hh   = (const float*)d_in[5];
    const float* b_ih   = (const float*)d_in[6];
    const float* b_hh   = (const float*)d_in[7];
    const float* W_out  = (const float*)d_in[8];
    const float* b_out  = (const float*)d_in[9];
    float* out = (float*)d_out;

    // workspace carve-up (256B aligned)
    char* ws = (char*)d_ws;
    size_t o = 0;
    auto carve = [&](size_t bytes) { size_t r = o; o = (o + bytes + 255) & ~(size_t)255; return r; };
    __bf16* embBf  = (__bf16*)(ws + carve((size_t)V  * E * 2));
    __bf16* WihBf  = (__bf16*)(ws + carve((size_t)G3 * E * 2));
    __bf16* WhhBf  = (__bf16*)(ws + carve((size_t)G3 * E * 2));
    __bf16* WoutBf = (__bf16*)(ws + carve((size_t)V  * E * 2));
    float*  h      = (float*) (ws + carve((size_t)Bsz * E * 4));
    __bf16* hbf    = (__bf16*)(ws + carve((size_t)Bsz * E * 2));
    float*  gi     = (float*) (ws + carve((size_t)Bsz * G3 * 4));
    float*  gh     = (float*) (ws + carve((size_t)Bsz * G3 * 4));
    int*    tok    = (int*)   (ws + carve((size_t)Bsz * 4));

    // one-time weight/embedding conversion to bf16 (L2-resident thereafter)
    {
        int n;
        n = V * E;   cvt_bf16_kernel<<<(n + 255) / 256, 256, 0, stream>>>(emb,   embBf,  n);
        n = G3 * E;  cvt_bf16_kernel<<<(n + 255) / 256, 256, 0, stream>>>(W_ih,  WihBf,  n);
        n = G3 * E;  cvt_bf16_kernel<<<(n + 255) / 256, 256, 0, stream>>>(W_hh,  WhhBf,  n);
        n = V * E;   cvt_bf16_kernel<<<(n + 255) / 256, 256, 0, stream>>>(W_out, WoutBf, n);
    }

    // h0 projection (full f32 precision, one-time) + tok init
    h0_kernel<<<(Bsz * E) / 256, 256, 0, stream>>>(latent, W_lat, b_lat, h, hbf, tok);

    // sequential greedy decode
    for (int t = 0; t < T; ++t) {
        gates_kernel<<<G3 / 32, 64, 0, stream>>>(embBf, tok, hbf, WihBf, WhhBf, b_ih, b_hh, gi, gh);
        update_kernel<<<(Bsz * E) / 256, 256, 0, stream>>>(gi, gh, h, hbf);
        outmm_kernel<<<V / 64, 64, 0, stream>>>(hbf, WoutBf, b_out, out, t);
        argmax_kernel<<<Bsz, 256, 0, stream>>>(out, t, tok);
    }
}
